// VectorQuantization_16758962389470
// MI455X (gfx1250) — compile-verified, matrix operand order checked by reading, not run on hardware
//
#include <hip/hip_runtime.h>
#include <hip/hip_bf16.h>
#include <math.h>

typedef __attribute__((ext_vector_type(16))) _Float16 v16h;
typedef __attribute__((ext_vector_type(8)))  _Float16 v8h;
typedef __attribute__((ext_vector_type(8)))  float    v8f;
typedef __attribute__((ext_vector_type(4)))  float    v4f;

#define NUM_EMB   1024
#define EMB_DIM   64
#define N_ROWS    65536          // 64 images * 1024 spatial positions
#define Q_ELEMS   4194304        // 64*64*32*32
#define SLICE     256            // codes staged in LDS per pass
#define LDSTRIDE  72             // halves per codebook row in LDS (padded: bank-conflict-free)
#define ROWTILES  4096           // N_ROWS / 16
#define TPB       256            // 8 waves per WG
#define WGITERS   256            // ROWTILES / (8 waves * 2 tiles)

// workspace byte offsets
#define WS_WF16   0              // 131072 B : codebook as f16
#define WS_ESQ    131072         // 4096 B   : ||e_k||^2 (f32)
#define WS_HIST   135168         // 4096 B   : histogram (u32)
#define WS_SSE    139264         // 16384 B  : per-rowtile SSE partials (f32)

__device__ __forceinline__ unsigned ldsOff(const void* p) {
  // low 32 bits of a flat LDS-aperture address are the LDS byte address
  return (unsigned)(__SIZE_TYPE__)p;
}

__device__ __forceinline__ void asyncLoadB128(unsigned loff, const void* gbase, unsigned goff) {
  asm volatile("global_load_async_to_lds_b128 %0, %1, %2 offset:0"
               :: "v"(loff), "v"(goff), "s"((unsigned long long)(__SIZE_TYPE__)gbase)
               : "memory");
}

// ---------------------------------------------------------------------------
// Kernel 0: codebook f32 -> f16, ||e||^2, zero histogram.  grid 256x256 = 65536
// ---------------------------------------------------------------------------
__global__ __launch_bounds__(256) void vq_prep(const float* __restrict__ w,
                                               _Float16* __restrict__ wf16,
                                               float* __restrict__ esq,
                                               unsigned* __restrict__ hist) {
  const int tid = blockIdx.x * 256 + threadIdx.x;
  wf16[tid] = (_Float16)w[tid];
  if (tid < NUM_EMB) {
    float s = 0.f;
#pragma unroll 8
    for (int d = 0; d < EMB_DIM; ++d) { const float v = w[tid * EMB_DIM + d]; s += v * v; }
    esq[tid] = s;
    hist[tid] = 0u;
  }
}

// ---------------------------------------------------------------------------
// Kernel 1: WMMA distance argmin + gather + q_out + partial reductions.
// grid 256 x 256 (8 waves/WG); wave -> 2 row tiles (32 rows): each LDS B-tile
// read feeds 4 WMMAs.  SSE computed from the winning score:
//   ||x - e||^2 = ||x||^2 - 2*(x.e - 0.5||e||^2) = sumx2 - 2*best
// Register budget kept < 256 VGPRs (avoids s_set_vgpr_msb in the hot loop).
// ---------------------------------------------------------------------------
__global__ __launch_bounds__(TPB) void vq_main(const float* __restrict__ x,
                                               const float* __restrict__ w,
                                               const _Float16* __restrict__ wf16,
                                               const float* __restrict__ esqg,
                                               unsigned* __restrict__ hist,
                                               float* __restrict__ ssep,
                                               float* __restrict__ qout,
                                               float* __restrict__ idxout) {
  __shared__ __align__(16) _Float16 wlds[SLICE * LDSTRIDE];   // 36,864 B
  __shared__ __align__(16) float    esql[NUM_EMB];            //  4,096 B
  __shared__ int idxs[8 * 32];                                //  1,024 B

  const int tid  = threadIdx.x;
  const int wv   = tid >> 5;       // wave in WG (0..7)
  const int lane = tid & 31;
  const int lm   = lane & 15;      // A row / B+C column within tile
  const int hh   = lane >> 4;      // half-wave selector

  // stage ||e||^2 into LDS (completion covered by first slice's asynccnt wait)
  asyncLoadB128(ldsOff(&esql[0]) + (unsigned)tid * 16u, esqg, (unsigned)tid * 16u);

  for (int g = blockIdx.x; g < WGITERS; g += gridDim.x) {
    const int rt0 = g * 16 + wv * 2;      // this wave's first row tile (of 2)
    size_t xb[2];
    v16h   a[2][2];
    float  sumx2[2];

    // ---- load 2x16 rows of x (transient f32), fold in sum(x^2), cvt to f16 A ----
#pragma unroll
    for (int t = 0; t < 2; ++t) {
      const int n0  = (rt0 + t) << 4;
      const int b   = n0 >> 10;
      const int hw0 = n0 & 1023;
      xb[t] = (size_t)b * 65536 + (size_t)hw0 + (size_t)lm;
      float s2 = 0.f;
#pragma unroll
      for (int blk = 0; blk < 4; ++blk)
#pragma unroll
        for (int j = 0; j < 8; ++j) {
          const int d = blk * 16 + hh * 8 + j;
          const float xv = x[xb[t] + (size_t)d * 1024];
          s2 += xv * xv;
          a[t][blk >> 1][(blk & 1) * 8 + j] = (_Float16)xv;
        }
      sumx2[t] = s2;
    }

    float best[2][8]; int bidx[2][8];
#pragma unroll
    for (int t = 0; t < 2; ++t)
#pragma unroll
      for (int v = 0; v < 8; ++v) { best[t][v] = -3.402823466e38f; bidx[t][v] = 0; }

    for (int s = 0; s < 4; ++s) {
      __syncthreads();                       // previous slice fully consumed
      // cooperative async copy: 256 codes * 128 B -> padded LDS (stride 144 B)
#pragma unroll
      for (int j = 0; j < 8; ++j) {
        const int e   = tid + j * TPB;       // [0, 2048) 16B transfers
        const int row = e >> 3, ir = e & 7;
        asyncLoadB128(ldsOff(&wlds[0]) + (unsigned)(row * (LDSTRIDE * 2) + ir * 16),
                      wf16,
                      (unsigned)((s * SLICE + row) * 128 + ir * 16));
      }
      asm volatile("s_wait_asynccnt 0" ::: "memory");
      __syncthreads();                       // slice visible to all waves

#pragma unroll 2
      for (int t8 = 0; t8 < 16; ++t8) {
        const int rr = t8 * 16 + lm;         // code row in slice (== B column's code)
        const int k  = s * SLICE + rr;       // global code id
        const v8h p0 = *(const v8h*)&wlds[rr * LDSTRIDE + hh * 8];
        const v8h p1 = *(const v8h*)&wlds[rr * LDSTRIDE + 16 + hh * 8];
        const v8h p2 = *(const v8h*)&wlds[rr * LDSTRIDE + 32 + hh * 8];
        const v8h p3 = *(const v8h*)&wlds[rr * LDSTRIDE + 48 + hh * 8];
        const v16h b0 = __builtin_shufflevector(p0, p1, 0,1,2,3,4,5,6,7,8,9,10,11,12,13,14,15);
        const v16h b1 = __builtin_shufflevector(p2, p3, 0,1,2,3,4,5,6,7,8,9,10,11,12,13,14,15);
        const float half_esq = 0.5f * esql[k];
#pragma unroll
        for (int t = 0; t < 2; ++t) {        // one B tile feeds both row tiles (4 WMMAs)
          v8f acc = {0.f, 0.f, 0.f, 0.f, 0.f, 0.f, 0.f, 0.f};
          acc = __builtin_amdgcn_wmma_f32_16x16x32_f16(false, a[t][0], false, b0, (short)0, acc, false, false);
          acc = __builtin_amdgcn_wmma_f32_16x16x32_f16(false, a[t][1], false, b1, (short)0, acc, false, false);
#pragma unroll
          for (int v = 0; v < 8; ++v) {      // score = x.e - 0.5||e||^2 ; argmax == argmin dist
            const float sc = acc[v] - half_esq;
            if (sc > best[t][v]) { best[t][v] = sc; bidx[t][v] = k; }
          }
        }
      }
    }

    // ---- argmax across the 16 lanes sharing each row (first-min tie-break) ----
#pragma unroll
    for (int off = 1; off < 16; off <<= 1)
#pragma unroll
      for (int t = 0; t < 2; ++t)
#pragma unroll
        for (int v = 0; v < 8; ++v) {
          const float ov = __shfl_xor(best[t][v], off, 32);
          const int   oi = __shfl_xor(bidx[t][v], off, 32);
          if (ov > best[t][v] || (ov == best[t][v] && oi < bidx[t][v])) {
            best[t][v] = ov; bidx[t][v] = oi;
          }
        }

    // exchange winners through LDS: row m = v + 8*hh within each tile
    if (lm == 0) {
#pragma unroll
      for (int t = 0; t < 2; ++t)
#pragma unroll
        for (int v = 0; v < 8; ++v) idxs[wv * 32 + t * 16 + hh * 8 + v] = bidx[t][v];
    }
    asm volatile("s_wait_dscnt 0" ::: "memory");  // same-wave LDS RAW fence

#pragma unroll
    for (int t = 0; t < 2; ++t) {
      // ---- SSE partial: sumx2 - 2*sum(best) over the tile's 16 rows ----
      float ssel = sumx2[t];
      if (lm == 0) {
        float bs = 0.f;
#pragma unroll
        for (int v = 0; v < 8; ++v) bs += best[t][v];   // lanes 0 & 16: rows 0-7 / 8-15
        ssel -= 2.f * bs;
      }
#pragma unroll
      for (int off = 1; off < 32; off <<= 1) ssel += __shfl_xor(ssel, off, 32);
      if (lane == 0) ssep[rt0 + t] = ssel;              // deterministic partial

      // ---- gather codebook row (exact f32) and write q_out (NCHW) ----
      const int myidx = idxs[wv * 32 + t * 16 + lm];
      const float* wrow = w + (size_t)myidx * EMB_DIM;
#pragma unroll
      for (int blk = 0; blk < 4; ++blk) {
        const v4f q0 = *(const v4f*)&wrow[blk * 16 + hh * 8];
        const v4f q1 = *(const v4f*)&wrow[blk * 16 + hh * 8 + 4];
#pragma unroll
        for (int j = 0; j < 4; ++j) {
          const int d = blk * 16 + hh * 8 + j;
          qout[xb[t] + (size_t)d * 1024]       = q0[j];
          qout[xb[t] + (size_t)(d + 4) * 1024] = q1[j];
        }
      }
      if (lane < 16) {
        const int ii = idxs[wv * 32 + t * 16 + lane];
        idxout[((rt0 + t) << 4) + lane] = (float)ii;
        atomicAdd(&hist[ii], 1u);                       // integer atomic: deterministic
      }
    }
  }
}

// ---------------------------------------------------------------------------
// Kernel 2: loss + perplexity.  1 block x 1024.
// ---------------------------------------------------------------------------
__global__ __launch_bounds__(1024) void vq_final(const unsigned* __restrict__ hist,
                                                 const float* __restrict__ ssep,
                                                 float* __restrict__ out) {
  __shared__ float shE[1024];
  __shared__ float shS[1024];
  const int t = threadIdx.x;
  const float p = (float)hist[t] * (1.0f / 65536.0f);
  shE[t] = -p * logf(p + 1e-10f);
  shS[t] = ssep[t] + ssep[t + 1024] + ssep[t + 2048] + ssep[t + 3072];
  __syncthreads();
  for (int o = 512; o > 0; o >>= 1) {
    if (t < o) { shE[t] += shE[t + o]; shS[t] += shS[t + o]; }
    __syncthreads();
  }
  if (t == 0) {
    out[Q_ELEMS]     = 1.25f * shS[0] * (1.0f / 4194304.0f);  // (1+0.25)*MSE
    out[Q_ELEMS + 1] = expf(shE[0]);
  }
}

// ---------------------------------------------------------------------------
extern "C" void kernel_launch(void* const* d_in, const int* in_sizes, int n_in,
                              void* d_out, int out_size, void* d_ws, size_t ws_size,
                              hipStream_t stream) {
  (void)in_sizes; (void)n_in; (void)out_size; (void)ws_size;
  const float* x = (const float*)d_in[0];       // (64,64,32,32) f32
  const float* w = (const float*)d_in[1];       // (1024,64) f32
  float* out = (float*)d_out;                   // q_out | loss | perplexity | idx
  char* ws = (char*)d_ws;
  _Float16* wf16 = (_Float16*)(ws + WS_WF16);
  float*    esq  = (float*)(ws + WS_ESQ);
  unsigned* hist = (unsigned*)(ws + WS_HIST);
  float*    ssep = (float*)(ws + WS_SSE);

  hipLaunchKernelGGL(vq_prep, dim3(256), dim3(256), 0, stream, w, wf16, esq, hist);
  hipLaunchKernelGGL(vq_main, dim3(256), dim3(TPB), 0, stream,
                     x, w, wf16, esq, hist, ssep, out, out + Q_ELEMS + 2);
  hipLaunchKernelGGL(vq_final, dim3(1), dim3(1024), 0, stream, hist, ssep, out);
}
